// RecurrentActorCritic_44504451121230
// MI455X (gfx1250) — matrix operand order repaired and checked by reading, not compiled, not run on hardware
//
#include <hip/hip_runtime.h>
#include <hip/hip_bf16.h>
#include <math.h>

typedef __attribute__((ext_vector_type(16))) _Float16 v16h;
typedef __attribute__((ext_vector_type(8)))  _Float16 v8h;
typedef __attribute__((ext_vector_type(8)))  float    v8f;

#define OBS 64
#define HID 64
#define HH  64
#define NA  16
#define TT  512
#define BB  1024
#define G3  192   // 3*H
#define NTILE (BB / 16)   // 64 batch tiles

#define LOG2E 1.4426950408889634f

// ---------- fast branchless activations (v_exp_f32 / v_rcp_f32) ----------

__device__ __forceinline__ float fast_sigmoid(float x) {
  return __builtin_amdgcn_rcpf(1.0f + __builtin_amdgcn_exp2f(-LOG2E * x));
}

__device__ __forceinline__ float fast_tanh(float x) {
  const float ax = __builtin_fabsf(x);
  const float t = __builtin_amdgcn_exp2f(-2.0f * LOG2E * ax);
  const float r = (1.0f - t) * __builtin_amdgcn_rcpf(1.0f + t);
  return __builtin_copysignf(r, x);
}

// ---------- fragment helpers (CDNA5 WMMA 16x16x32 f16 layouts) ----------

__device__ __forceinline__ v8f wmma_f16(v16h a, v16h b, v8f c) {
  return __builtin_amdgcn_wmma_f32_16x16x32_f16(false, a, false, b, (short)0, c,
                                                false, false);
}

__device__ __forceinline__ int a_k(int e, int hi) {
  // contiguous k order used here for A (load_b_w matches): lanes 0-15 K={0..15}, lanes 16-31 K={16..31}
  return e + (hi << 4);
}

// load A fragment (16 rows x 32 K) from row-major f32 matrix
__device__ __forceinline__ v16h load_a_f32(const float* __restrict__ src, long ld,
                                           long r0, int k0, int lane) {
  v16h a;
  const int m = lane & 15, hi = lane >> 4;
  const float* row = src + (r0 + m) * ld + k0;
#pragma unroll
  for (int e = 0; e < 16; ++e) a[e] = (_Float16)row[a_k(e, hi)];
  return a;
}

// load A fragment from a row-major f16 LDS tile
__device__ __forceinline__ v16h load_a_h(const _Float16* __restrict__ src, int ld,
                                         int k0, int lane) {
  v16h a;
  const int m = lane & 15, hi = lane >> 4;
  const _Float16* row = src + m * ld + k0;
#pragma unroll
  for (int e = 0; e < 16; ++e) a[e] = row[a_k(e, hi)];
  return a;
}

// load B fragment (32 K x 16 N) from f16 weights stored [out_ch][K] row-major
__device__ __forceinline__ v16h load_b_w(const _Float16* __restrict__ W, int ldW,
                                         int n0, int k0, int lane) {
  v16h b;
  const int n = lane & 15, hi = lane >> 4;
  const _Float16* col = W + (n0 + n) * ldW + k0 + (hi << 4);
#pragma unroll
  for (int e = 0; e < 16; ++e) b[e] = col[e];
  return b;
}

// Same-wave LDS RAW/WAR is safe without a block barrier (LDS ops from one wave
// are processed in order, DScnt in order); this is just a compiler fence.
__device__ __forceinline__ void wave_fence() { __builtin_amdgcn_wave_barrier(); }

// ---------- phase 1: gx = tanh(x@Wb^T + bb) @ Wih^T + bih  (parallel T*B) ----------
// gxs layout: [tile][lane][jt*8+r] f16, tile = t*NTILE + b0/16 (lane-swizzled C/D)

#define P1W 8
__global__ __launch_bounds__(32 * P1W)
void k_phase1(const float* __restrict__ x, const float* __restrict__ Wb,
              const float* __restrict__ bb, const float* __restrict__ Wih,
              const float* __restrict__ bih, _Float16* __restrict__ gxs) {
  __shared__ _Float16 sWb[HID * OBS];
  __shared__ _Float16 sWih[G3 * HID];
  __shared__ _Float16 sbuf[P1W][16 * 72];

  const int tid = threadIdx.x;
  for (int i = tid; i < HID * OBS; i += blockDim.x) sWb[i] = (_Float16)Wb[i];
  for (int i = tid; i < G3 * HID; i += blockDim.x) sWih[i] = (_Float16)Wih[i];
  __syncthreads();

  const int lane = tid & 31, w = tid >> 5;
  const long tile = (long)blockIdx.x * P1W + w;   // tile of 16 rows in [0, T*B/16)
  const long r0 = tile * 16;
  const int n = lane & 15, hi = lane >> 4;
  _Float16* buf = &sbuf[w][0];

  // GEMM1: xb = tanh(x @ Wb^T + bb)
  v16h xa0 = load_a_f32(x, OBS, r0, 0, lane);
  v16h xa1 = load_a_f32(x, OBS, r0, 32, lane);
#pragma unroll
  for (int ct = 0; ct < 4; ++ct) {
    const float bv = bb[ct * 16 + n];
    v8f acc;
#pragma unroll
    for (int r = 0; r < 8; ++r) acc[r] = bv;
    acc = wmma_f16(xa0, load_b_w(sWb, OBS, ct * 16, 0, lane), acc);
    acc = wmma_f16(xa1, load_b_w(sWb, OBS, ct * 16, 32, lane), acc);
#pragma unroll
    for (int r = 0; r < 8; ++r)
      buf[(r + 8 * hi) * 72 + ct * 16 + n] = (_Float16)fast_tanh(acc[r]);
  }
  wave_fence();  // per-wave buffer: same-wave LDS ordering suffices
  v16h ba0 = load_a_h(buf, 72, 0, lane);
  v16h ba1 = load_a_h(buf, 72, 32, lane);

  // GEMM2: gx = xb @ Wih^T + bih -> swizzled f16 workspace, 12 x b128 per lane
  v8h* gout = (v8h*)(gxs + ((size_t)tile * 32 + lane) * 96);
#pragma unroll
  for (int jt = 0; jt < 12; ++jt) {
    const float bv = bih[jt * 16 + n];
    v8f acc;
#pragma unroll
    for (int r = 0; r < 8; ++r) acc[r] = bv;
    acc = wmma_f16(ba0, load_b_w(sWih, HID, jt * 16, 0, lane), acc);
    acc = wmma_f16(ba1, load_b_w(sWih, HID, jt * 16, 32, lane), acc);
    v8h o;
#pragma unroll
    for (int r = 0; r < 8; ++r) o[r] = (_Float16)acc[r];
    gout[jt] = o;
  }
}

// ---------- phase 2: sequential GRU scan (parallel over B only) ----------
// outs_s layout: [tile][lane][ct*8+r] f16 (lane-swizzled C/D)

#define P2W 4
__global__ __launch_bounds__(32 * P2W)
void k_phase2(const float* __restrict__ hx, const unsigned char* __restrict__ dones,
              const float* __restrict__ Whh, const float* __restrict__ bhh,
              const _Float16* __restrict__ gxs, _Float16* __restrict__ outs_s,
              float* __restrict__ hT) {
  __shared__ _Float16 sW[G3 * HH];
  __shared__ _Float16 sbuf[P2W][16 * 72];

  const int tid = threadIdx.x;
  for (int i = tid; i < G3 * HH; i += blockDim.x) sW[i] = (_Float16)Whh[i];
  __syncthreads();

  const int lane = tid & 31, w = tid >> 5;
  const int bt = blockIdx.x * P2W + w;       // batch tile [0, NTILE)
  const int b0 = bt * 16;
  const int n = lane & 15, hi = lane >> 4;
  _Float16* buf = &sbuf[w][0];

  // h0 in both layouts
  v16h hA0 = load_a_f32(hx, HH, b0, 0, lane);
  v16h hA1 = load_a_f32(hx, HH, b0, 32, lane);
  v8f hcd[4];
#pragma unroll
  for (int ct = 0; ct < 4; ++ct)
#pragma unroll
    for (int r = 0; r < 8; ++r)
      hcd[ct][r] = hx[(b0 + r + 8 * hi) * HH + ct * 16 + n];

  float bh[12];
#pragma unroll
  for (int jt = 0; jt < 12; ++jt) bh[jt] = bhh[jt * 16 + n];

  for (int t = 0; t < TT; ++t) {
    // vectorized done masks: one b64 per lane + one byte for the A-side
    const unsigned char* dnb = dones + (size_t)t * BB + b0;
    const uint2 dp = *(const uint2*)(dnb + 8 * hi);
    const _Float16 mA = dnb[lane & 15] ? (_Float16)0.0f : (_Float16)1.0f;
    float mr[8];
#pragma unroll
    for (int r = 0; r < 8; ++r) {
      const unsigned v = (r < 4) ? (dp.x >> (8 * r)) : (dp.y >> (8 * (r - 4)));
      mr[r] = (v & 0xFFu) ? 0.0f : 1.0f;
    }
#pragma unroll
    for (int e = 0; e < 16; ++e) { hA0[e] *= mA; hA1[e] *= mA; }
#pragma unroll
    for (int ct = 0; ct < 4; ++ct)
#pragma unroll
      for (int r = 0; r < 8; ++r) hcd[ct][r] *= mr[r];

    // per-step gate inputs: 12 x global_load_b128, lane-contiguous
    const v8h* gxl = (const v8h*)(gxs + (((size_t)t * NTILE + bt) * 32 + lane) * 96);
    v8h gxv[12];
#pragma unroll
    for (int j = 0; j < 12; ++j) gxv[j] = gxl[j];

    // gh = h @ Whh^T + bhh  (12 N-tiles x 2 K-steps = 24 WMMAs)
    v8f acc[12];
#pragma unroll
    for (int jt = 0; jt < 12; ++jt) {
#pragma unroll
      for (int r = 0; r < 8; ++r) acc[jt][r] = bh[jt];
      acc[jt] = wmma_f16(hA0, load_b_w(sW, HH, jt * 16, 0, lane), acc[jt]);
      acc[jt] = wmma_f16(hA1, load_b_w(sW, HH, jt * 16, 32, lane), acc[jt]);
    }

    // prefetch next step's swizzled gate block (global_prefetch_b8)
    if (t + 1 < TT)
      __builtin_prefetch(gxs + (((size_t)(t + 1) * NTILE + bt) * 32 + lane) * 96, 0, 1);

    v8h oh[4];
#pragma unroll
    for (int ct = 0; ct < 4; ++ct) {
#pragma unroll
      for (int r = 0; r < 8; ++r) {
        const int m = r + 8 * hi;
        const int c = ct * 16 + n;
        const float rg = fast_sigmoid((float)gxv[ct][r] + acc[ct][r]);
        const float zg = fast_sigmoid((float)gxv[ct + 4][r] + acc[ct + 4][r]);
        const float ng = fast_tanh((float)gxv[ct + 8][r] + rg * acc[ct + 8][r]);
        const float hn = (1.0f - zg) * ng + zg * hcd[ct][r];
        hcd[ct][r] = hn;
        const _Float16 hh16 = (_Float16)hn;
        oh[ct][r] = hh16;
        buf[m * 72 + c] = hh16;  // stage for C/D -> A transpose
      }
    }
    // outs: 4 x b128 per lane, lane-contiguous swizzled layout
    v8h* oout = (v8h*)(outs_s + (((size_t)t * NTILE + bt) * 32 + lane) * 32);
#pragma unroll
    for (int ct = 0; ct < 4; ++ct) oout[ct] = oh[ct];

    // per-wave transpose buffer: same-wave in-order LDS makes barriers unnecessary
    wave_fence();
    hA0 = load_a_h(buf, 72, 0, lane);
    hA1 = load_a_h(buf, 72, 32, lane);
    wave_fence();
  }

  // hT = final hidden state (f32 from registers)
#pragma unroll
  for (int ct = 0; ct < 4; ++ct)
#pragma unroll
    for (int r = 0; r < 8; ++r)
      hT[(b0 + r + 8 * hi) * HH + ct * 16 + n] = hcd[ct][r];
}

// ---------- phase 3: actor/critic heads (parallel T*B) ----------

#define P3W 8
__global__ __launch_bounds__(32 * P3W)
void k_phase3(const _Float16* __restrict__ outs_s, const float* __restrict__ Wa1,
              const float* __restrict__ ba1, const float* __restrict__ Wa2,
              const float* __restrict__ ba2, const float* __restrict__ Wc1,
              const float* __restrict__ bc1, const float* __restrict__ Wc2,
              const float* __restrict__ bc2, float* __restrict__ logits,
              float* __restrict__ values) {
  __shared__ _Float16 sWa1[HID * HH];
  __shared__ _Float16 sWa2[NA * HID];
  __shared__ _Float16 sWc1[HID * HH];
  __shared__ _Float16 sWc2[HID];
  __shared__ _Float16 sbuf[P3W][16 * 72];

  const int tid = threadIdx.x;
  for (int i = tid; i < HID * HH; i += blockDim.x) {
    sWa1[i] = (_Float16)Wa1[i];
    sWc1[i] = (_Float16)Wc1[i];
  }
  for (int i = tid; i < NA * HID; i += blockDim.x) sWa2[i] = (_Float16)Wa2[i];
  for (int i = tid; i < HID; i += blockDim.x) sWc2[i] = (_Float16)Wc2[i];
  __syncthreads();

  const int lane = tid & 31, w = tid >> 5;
  const long tile = (long)blockIdx.x * P3W + w;
  const long r0 = tile * 16;
  const int n = lane & 15, hi = lane >> 4;
  _Float16* buf = &sbuf[w][0];

  // load swizzled outs (4 x b128 per lane), transpose via LDS into A fragments
  const v8h* oin = (const v8h*)(outs_s + ((size_t)tile * 32 + lane) * 32);
  v8h ov[4];
#pragma unroll
  for (int ct = 0; ct < 4; ++ct) ov[ct] = oin[ct];
#pragma unroll
  for (int ct = 0; ct < 4; ++ct)
#pragma unroll
    for (int r = 0; r < 8; ++r)
      buf[(r + 8 * hi) * 72 + ct * 16 + n] = ov[ct][r];
  wave_fence();
  v16h oa0 = load_a_h(buf, 72, 0, lane);
  v16h oa1 = load_a_h(buf, 72, 32, lane);
  wave_fence();

  // actor hidden: a1 = tanh(outs @ Wa1^T + ba1)
#pragma unroll
  for (int ct = 0; ct < 4; ++ct) {
    const float bv = ba1[ct * 16 + n];
    v8f acc;
#pragma unroll
    for (int r = 0; r < 8; ++r) acc[r] = bv;
    acc = wmma_f16(oa0, load_b_w(sWa1, HH, ct * 16, 0, lane), acc);
    acc = wmma_f16(oa1, load_b_w(sWa1, HH, ct * 16, 32, lane), acc);
#pragma unroll
    for (int r = 0; r < 8; ++r)
      buf[(r + 8 * hi) * 72 + ct * 16 + n] = (_Float16)fast_tanh(acc[r]);
  }
  wave_fence();
  v16h a0 = load_a_h(buf, 72, 0, lane);
  v16h a1 = load_a_h(buf, 72, 32, lane);
  wave_fence();

  // logits = a1 @ Wa2^T + ba2  (single 16-wide N tile)
  {
    const float bv = ba2[n];
    v8f acc;
#pragma unroll
    for (int r = 0; r < 8; ++r) acc[r] = bv;
    acc = wmma_f16(a0, load_b_w(sWa2, HID, 0, 0, lane), acc);
    acc = wmma_f16(a1, load_b_w(sWa2, HID, 0, 32, lane), acc);
#pragma unroll
    for (int r = 0; r < 8; ++r) logits[(r0 + r + 8 * hi) * NA + n] = acc[r];
  }

  // critic hidden: c1 = tanh(outs @ Wc1^T + bc1)
#pragma unroll
  for (int ct = 0; ct < 4; ++ct) {
    const float bv = bc1[ct * 16 + n];
    v8f acc;
#pragma unroll
    for (int r = 0; r < 8; ++r) acc[r] = bv;
    acc = wmma_f16(oa0, load_b_w(sWc1, HH, ct * 16, 0, lane), acc);
    acc = wmma_f16(oa1, load_b_w(sWc1, HH, ct * 16, 32, lane), acc);
#pragma unroll
    for (int r = 0; r < 8; ++r)
      buf[(r + 8 * hi) * 72 + ct * 16 + n] = (_Float16)fast_tanh(acc[r]);
  }
  wave_fence();
  v16h c0 = load_a_h(buf, 72, 0, lane);
  v16h c1 = load_a_h(buf, 72, 32, lane);

  // values = c1 @ Wc2^T + bc2 : B matrix has only column n==0 populated
  {
    v8f acc = {};
    v16h b0f, b1f;
#pragma unroll
    for (int e = 0; e < 16; ++e) {
      const int k = e + (hi << 4);
      b0f[e] = (n == 0) ? sWc2[k] : (_Float16)0.0f;
      b1f[e] = (n == 0) ? sWc2[32 + k] : (_Float16)0.0f;
    }
    acc = wmma_f16(c0, b0f, acc);
    acc = wmma_f16(c1, b1f, acc);
    if (n == 0) {
      const float bv = bc2[0];
#pragma unroll
      for (int r = 0; r < 8; ++r) values[r0 + r + 8 * hi] = acc[r] + bv;
    }
  }
}

// ---------- launch ----------

extern "C" void kernel_launch(void* const* d_in, const int* in_sizes, int n_in,
                              void* d_out, int out_size, void* d_ws, size_t ws_size,
                              hipStream_t stream) {
  (void)in_sizes; (void)n_in; (void)out_size; (void)ws_size;
  const float* x   = (const float*)d_in[0];
  const float* hx  = (const float*)d_in[1];
  const unsigned char* dones = (const unsigned char*)d_in[2];  // jnp bool = 1 byte
  const float* Wb  = (const float*)d_in[3];
  const float* bb  = (const float*)d_in[4];
  const float* Wih = (const float*)d_in[5];
  const float* Whh = (const float*)d_in[6];
  const float* bih = (const float*)d_in[7];
  const float* bhh = (const float*)d_in[8];
  const float* Wa1 = (const float*)d_in[9];
  const float* ba1 = (const float*)d_in[10];
  const float* Wa2 = (const float*)d_in[11];
  const float* ba2 = (const float*)d_in[12];
  const float* Wc1 = (const float*)d_in[13];
  const float* bc1 = (const float*)d_in[14];
  const float* Wc2 = (const float*)d_in[15];
  const float* bc2 = (const float*)d_in[16];

  float* logits = (float*)d_out;                       // (T,B,A)
  float* values = logits + (size_t)TT * BB * NA;       // (T,B)
  float* hT     = values + (size_t)TT * BB;            // (1,B,H)

  _Float16* gxs    = (_Float16*)d_ws;                  // swizzled (T*B/16, 32, 96) f16
  _Float16* outs_s = (_Float16*)((char*)d_ws + (size_t)TT * BB * G3 * sizeof(_Float16));

  const long rows = (long)TT * BB;                     // 524288
  k_phase1<<<rows / (16 * P1W), 32 * P1W, 0, stream>>>(x, Wb, bb, Wih, bih, gxs);
  k_phase2<<<NTILE / P2W, 32 * P2W, 0, stream>>>(hx, dones, Whh, bhh, gxs, outs_s, hT);
  k_phase3<<<rows / (16 * P3W), 32 * P3W, 0, stream>>>(outs_s, Wa1, ba1, Wa2, ba2,
                                                       Wc1, bc1, Wc2, bc2, logits, values);
}
